// SparseMHADecoder_76759655514199
// MI455X (gfx1250) — matrix operand order repaired and checked by reading, not compiled, not run on hardware
//
#include <hip/hip_runtime.h>
#include <hip/hip_bf16.h>

// ---------------- problem constants (fixed by the reference setup) ----------
#define BB     2
#define LQ     2048
#define LKV    1024
#define DIN    512          // DQ = DK = DV_IN
#define HNUM   8
#define DQK    64
#define DVH    64
#define DOUT   512
#define SPAN   128
#define STRIDE 2
#define WIDTH  ((SPAN + STRIDE - 1) / STRIDE)   // 64

typedef __attribute__((ext_vector_type(16))) _Float16 v16h;
typedef __attribute__((ext_vector_type(8)))  _Float16 v8h;
typedef __attribute__((ext_vector_type(8)))  float    v8f;

// ---------------------------------------------------------------------------
// One-time elementwise fp32 -> f16 convert (n divisible by 1024).
// ---------------------------------------------------------------------------
__global__ __launch_bounds__(256)
void cvt_f32_f16_kernel(const float* __restrict__ src,
                        _Float16* __restrict__ dst, int n)
{
    const int i = (blockIdx.x * 256 + threadIdx.x) * 4;
    if (i < n) {
        const float4 v = *(const float4*)(src + i);
        dst[i]     = (_Float16)v.x;
        dst[i + 1] = (_Float16)v.y;
        dst[i + 2] = (_Float16)v.z;
        dst[i + 3] = (_Float16)v.w;
    }
}

// ---------------------------------------------------------------------------
// One-time weight transpose + convert: Wt[n][k] = (f16) W[k][n].
// Reads coalesced; K-contiguous f16 rows let the GEMM load B fragments as
// two b128 loads per 16x16 tile per K-step.
// ---------------------------------------------------------------------------
template <int NN, int KK>
__global__ __launch_bounds__(256)
void transpose_f32_f16_kernel(const float* __restrict__ W,
                              _Float16* __restrict__ Wt)
{
    const int tid = blockIdx.x * 256 + threadIdx.x;   // = k*NN + n
    if (tid < NN * KK) {
        const int k = tid / NN, n = tid % NN;         // NN is a power of two
        Wt[(size_t)n * KK + k] = (_Float16)W[tid];
    }
}

// ---------------------------------------------------------------------------
// f16-in / fp32-out GEMM: C[M,N] = A[M,K] @ W[K,N] with A f16 row-major and
// W pre-transposed to f16 (NN x KK).  One wave owns a 16x64 C strip
// (4 accumulators); per K-step of 32 the wave issues 2 b128 loads for A and
// 8 b128 loads for B, then 4 v_wmma_f32_16x16x32_f16.  The K loop is
// explicitly double-buffered (next step's fragments are loaded before the
// current step's WMMAs) so loads overlap the matrix pipeline.  MODE==1
// remaps the C write into head-major (B, H, 1<<LSHIFT, 64) layout.
// ---------------------------------------------------------------------------
template <int MODE, int LSHIFT, int NN, int KK>
__global__ __launch_bounds__(256)
void gemm_wmma_kernel(const _Float16* __restrict__ A,   // M x KK row-major
                      const _Float16* __restrict__ Wt,  // NN x KK (transposed W)
                      float* __restrict__ C,
                      int M)
{
    const int lane  = threadIdx.x & 31;
    const int wave  = threadIdx.x >> 5;
    const int n0    = (blockIdx.x * 8 + wave) * 64;   // this wave's 64-col strip
    const int mtile = blockIdx.y;
    if (n0 >= NN) return;

    const int half8 = lane >> 4;                      // 0: lanes 0-15, 1: 16-31
    const int mrow  = mtile * 16 + (lane & 15);       // A row for lane's frag
    const int ncol  = n0 + (lane & 15);               // first B/C column

    // A fragment = run [k0+half8*16, +16): two b128.
    const _Float16* __restrict__ arow = A + (size_t)mrow * KK + (half8 << 4);
    // B fragment for N-tile nt = Wt[ncol+16*nt][k0+half8*16 .. +16): two b128.
    const _Float16* __restrict__ wrow = Wt + (size_t)ncol * KK + (half8 << 4);

    struct Frag { v16h a, b0, b1, b2, b3; };
    auto ldfrag = [&](int k0) {
        Frag f;
        const v8h a0  = *(const v8h*)(arow + k0);
        const v8h a1  = *(const v8h*)(arow + k0 + 8);
        const v8h b00 = *(const v8h*)(wrow + k0);
        const v8h b01 = *(const v8h*)(wrow + k0 + 8);
        const v8h b10 = *(const v8h*)(wrow + 16 * KK + k0);
        const v8h b11 = *(const v8h*)(wrow + 16 * KK + k0 + 8);
        const v8h b20 = *(const v8h*)(wrow + 32 * KK + k0);
        const v8h b21 = *(const v8h*)(wrow + 32 * KK + k0 + 8);
        const v8h b30 = *(const v8h*)(wrow + 48 * KK + k0);
        const v8h b31 = *(const v8h*)(wrow + 48 * KK + k0 + 8);
        f.a  = __builtin_shufflevector(a0, a1,
            0, 1, 2, 3, 4, 5, 6, 7, 8, 9, 10, 11, 12, 13, 14, 15);
        f.b0 = __builtin_shufflevector(b00, b01,
            0, 1, 2, 3, 4, 5, 6, 7, 8, 9, 10, 11, 12, 13, 14, 15);
        f.b1 = __builtin_shufflevector(b10, b11,
            0, 1, 2, 3, 4, 5, 6, 7, 8, 9, 10, 11, 12, 13, 14, 15);
        f.b2 = __builtin_shufflevector(b20, b21,
            0, 1, 2, 3, 4, 5, 6, 7, 8, 9, 10, 11, 12, 13, 14, 15);
        f.b3 = __builtin_shufflevector(b30, b31,
            0, 1, 2, 3, 4, 5, 6, 7, 8, 9, 10, 11, 12, 13, 14, 15);
        return f;
    };

    v8f acc0 = {}, acc1 = {}, acc2 = {}, acc3 = {};

    Frag cur = ldfrag(0);
    #pragma unroll 2
    for (int k0 = 0; k0 < KK - 32; k0 += 32) {
        __builtin_prefetch(arow + k0 + 160, 0, 1);    // safe: ws regions follow
        Frag nxt = ldfrag(k0 + 32);                   // overlaps the WMMAs below
        acc0 = __builtin_amdgcn_wmma_f32_16x16x32_f16(false, cur.a, false, cur.b0,
                                                      (short)0, acc0, false, false);
        acc1 = __builtin_amdgcn_wmma_f32_16x16x32_f16(false, cur.a, false, cur.b1,
                                                      (short)0, acc1, false, false);
        acc2 = __builtin_amdgcn_wmma_f32_16x16x32_f16(false, cur.a, false, cur.b2,
                                                      (short)0, acc2, false, false);
        acc3 = __builtin_amdgcn_wmma_f32_16x16x32_f16(false, cur.a, false, cur.b3,
                                                      (short)0, acc3, false, false);
        cur = nxt;
    }
    // peeled last step (no next-fragment load, no branch in the loop body)
    acc0 = __builtin_amdgcn_wmma_f32_16x16x32_f16(false, cur.a, false, cur.b0,
                                                  (short)0, acc0, false, false);
    acc1 = __builtin_amdgcn_wmma_f32_16x16x32_f16(false, cur.a, false, cur.b1,
                                                  (short)0, acc1, false, false);
    acc2 = __builtin_amdgcn_wmma_f32_16x16x32_f16(false, cur.a, false, cur.b2,
                                                  (short)0, acc2, false, false);
    acc3 = __builtin_amdgcn_wmma_f32_16x16x32_f16(false, cur.a, false, cur.b3,
                                                  (short)0, acc3, false, false);

    // C/D layout: VGPR r holds row (mtile*16 + r + 8*half8), col = ncol+16*nt.
    const v8f accs[4] = {acc0, acc1, acc2, acc3};
    #pragma unroll
    for (int nt = 0; nt < 4; ++nt) {
        const int col = ncol + nt * 16;
        #pragma unroll
        for (int r = 0; r < 8; ++r) {
            const int row = mtile * 16 + r + (half8 << 3);
            if (MODE == 0) {
                C[(size_t)row * NN + col] = accs[nt][r];
            } else {
                const int bb = row >> LSHIFT;
                const int jj = row & ((1 << LSHIFT) - 1);
                const int hh = col >> 6, dd = col & 63;
                const int Hh = NN >> 6;
                C[(((size_t)(bb * Hh + hh)) << LSHIFT) * 64 +
                  ((size_t)jj * 64) + dd] = accs[nt][r];
            }
        }
    }
}

// ---------------------------------------------------------------------------
// Banded local attention core.  Inputs are head-major fp32:
//   Qp: (B, H, LQ,  DQK)   Kp: (B, H, LKV, DQK)   Vp: (B, H, LKV, DVH)
// Output AO is f16, standard (B*LQ, H*DVH) layout, feeding the final GEMM
// directly.  One wave per query; lane l owns window slots {l, l+32} for the
// scores and output channels {2l, 2l+1} for the accumulation.
// ---------------------------------------------------------------------------
__global__ __launch_bounds__(256)
void attn_kernel(const float* __restrict__ Qp,
                 const float* __restrict__ Kp,
                 const float* __restrict__ Vp,
                 _Float16* __restrict__ AO)
{
    const int tid  = threadIdx.x;
    const int wave = tid >> 5;
    const int lane = tid & 31;
    const int gq   = blockIdx.x * 8 + wave;          // over B*H*LQ
    const int j    = gq % LQ;
    const int h    = (gq / LQ) % HNUM;
    const int b    = gq / (LQ * HNUM);

    __shared__ float s_q[8][DQK];
    __shared__ float s_p[8][WIDTH];

    // stage this wave's Q row (64 floats, 2 per lane)
    const float* qrow = Qp + (((size_t)(b * HNUM + h)) * LQ + j) * DQK;
    s_q[wave][2 * lane]     = qrow[2 * lane];
    s_q[wave][2 * lane + 1] = qrow[2 * lane + 1];
    __syncthreads();

    const float* Kb = Kp + ((size_t)(b * HNUM + h)) * LKV * DQK;
    const float* Vb = Vp + ((size_t)(b * HNUM + h)) * LKV * DVH;
    const int base  = j / STRIDE - (WIDTH - 1);

    // ---- scores: two window slots per lane --------------------------------
    float sc[2];
    #pragma unroll
    for (int t = 0; t < 2; ++t) {
        const int w   = lane + t * 32;
        const int col = base + w;
        const bool valid = (col >= 0) && (col < LKV) &&
                           ((j % STRIDE) <= SPAN - 1 - (WIDTH - 1 - w) * STRIDE);
        float s = -1e30f;
        if (valid) {
            const float* krow = Kb + (size_t)col * DQK;
            float d0 = 0.f;
            #pragma unroll
            for (int d = 0; d < DQK; d += 4) {
                const float4 kv = *(const float4*)(krow + d);
                d0 += s_q[wave][d + 0] * kv.x + s_q[wave][d + 1] * kv.y +
                      s_q[wave][d + 2] * kv.z + s_q[wave][d + 3] * kv.w;
            }
            s = d0;
        }
        sc[t] = s;
    }

    // ---- softmax over the 64 window slots (wave-wide reductions) ----------
    float mx = fmaxf(sc[0], sc[1]);
    #pragma unroll
    for (int off = 16; off > 0; off >>= 1)
        mx = fmaxf(mx, __shfl_xor(mx, off, 32));
    const float e0 = __expf(sc[0] - mx);
    const float e1 = __expf(sc[1] - mx);
    float sum = e0 + e1;
    #pragma unroll
    for (int off = 16; off > 0; off >>= 1)
        sum += __shfl_xor(sum, off, 32);
    const float inv = 1.0f / sum;
    s_p[wave][lane]      = e0 * inv;
    s_p[wave][lane + 32] = e1 * inv;
    __syncthreads();

    // ---- output: lane owns channels {2*lane, 2*lane+1} --------------------
    float ax = 0.f, ay = 0.f;
    const int dch = 2 * lane;
    #pragma unroll 4
    for (int w = 0; w < WIDTH; ++w) {
        int col = base + w;
        col = col < 0 ? 0 : (col > LKV - 1 ? LKV - 1 : col);  // clip; p==0 when invalid
        const float  p  = s_p[wave][w];
        const float2 vv = *(const float2*)(Vb + (size_t)col * DVH + dch);
        ax += p * vv.x;
        ay += p * vv.y;
    }
    _Float16* orow = AO + ((size_t)b * LQ + j) * (HNUM * DVH) + h * DVH + dch;
    orow[0] = (_Float16)ax;
    orow[1] = (_Float16)ay;
}

// ---------------------------------------------------------------------------
extern "C" void kernel_launch(void* const* d_in, const int* in_sizes, int n_in,
                              void* d_out, int out_size, void* d_ws, size_t ws_size,
                              hipStream_t stream)
{
    const float* q    = (const float*)d_in[0];
    const float* k    = (const float*)d_in[1];
    const float* v    = (const float*)d_in[2];
    const float* Wq   = (const float*)d_in[3];
    const float* Wk   = (const float*)d_in[4];
    const float* Wv   = (const float*)d_in[5];
    const float* Wout = (const float*)d_in[6];
    float*       out  = (float*)d_out;

    // workspace layout: fp32 head-major Q/K/V, then f16 staging buffers
    const size_t nQ  = (size_t)BB * LQ  * DIN;   // 2,097,152
    const size_t nKV = (size_t)BB * LKV * DIN;   // 1,048,576
    const size_t nW  = (size_t)DIN * DIN;        //   262,144

    float* Qp = (float*)d_ws;
    float* Kp = Qp + nQ;
    float* Vp = Kp + nKV;

    _Float16* qh  = (_Float16*)(Vp + nKV);
    _Float16* kh  = qh  + nQ;
    _Float16* vh  = kh  + nKV;
    _Float16* AOh = vh  + nKV;
    _Float16* Wqt = AOh + nQ;
    _Float16* Wkt = Wqt + nW;
    _Float16* Wvt = Wkt + nW;
    _Float16* Wot = Wvt + nW;

    const dim3 blk(256);

    // ---- one-time packs: activations -> f16 rows, weights -> f16 transposed
    cvt_f32_f16_kernel<<<dim3(nQ  / 1024), blk, 0, stream>>>(q, qh, (int)nQ);
    cvt_f32_f16_kernel<<<dim3(nKV / 1024), blk, 0, stream>>>(k, kh, (int)nKV);
    cvt_f32_f16_kernel<<<dim3(nKV / 1024), blk, 0, stream>>>(v, vh, (int)nKV);
    transpose_f32_f16_kernel<512, 512><<<dim3(nW / 256), blk, 0, stream>>>(Wq,   Wqt);
    transpose_f32_f16_kernel<512, 512><<<dim3(nW / 256), blk, 0, stream>>>(Wk,   Wkt);
    transpose_f32_f16_kernel<512, 512><<<dim3(nW / 256), blk, 0, stream>>>(Wv,   Wvt);
    transpose_f32_f16_kernel<512, 512><<<dim3(nW / 256), blk, 0, stream>>>(Wout, Wot);

    // ---- projections (head-major fp32 outputs; Lrows 2048/1024 -> 11/10)
    gemm_wmma_kernel<1, 11, 512, 512><<<dim3(1, (BB * LQ)  / 16), blk, 0, stream>>>(
        qh, Wqt, Qp, BB * LQ);
    gemm_wmma_kernel<1, 10, 512, 512><<<dim3(1, (BB * LKV) / 16), blk, 0, stream>>>(
        kh, Wkt, Kp, BB * LKV);
    gemm_wmma_kernel<1, 10, 512, 512><<<dim3(1, (BB * LKV) / 16), blk, 0, stream>>>(
        vh, Wvt, Vp, BB * LKV);

    // ---- banded local attention (writes f16 AO)
    attn_kernel<<<dim3((BB * HNUM * LQ) / 8), blk, 0, stream>>>(Qp, Kp, Vp, AOh);

    // ---- output projection (standard fp32 layout)
    gemm_wmma_kernel<0, 0, 512, 512><<<dim3(1, (BB * LQ) / 16), blk, 0, stream>>>(
        AOh, Wot, out, BB * LQ);
}
